// Encoder_39676907880550
// MI455X (gfx1250) — matrix operand order repaired and checked by reading, not compiled, not run on hardware
//
#include <hip/hip_runtime.h>
#include <cstdint>

// ---------------------------------------------------------------------------
// MI455X / gfx1250 transformer encoder, bf16 WMMA (f32 accumulate) everywhere.
// B=8 T=1024 D_IN=320 L=6 H=8 DK=DV=64 DM=512 DI=2048
// ---------------------------------------------------------------------------

typedef __attribute__((ext_vector_type(16))) __bf16 v16bf;
typedef __attribute__((ext_vector_type(8)))  float  v8f;

union AFrag { v16bf v; uint4 q[2]; };

// gfx1250 async global->LDS copy (GV mode, 16 bytes per lane), tracked by ASYNCcnt
__device__ __forceinline__ void async_copy_b128(void* lds_ptr, const void* gptr) {
  uint32_t lds_off = (uint32_t)(uintptr_t)lds_ptr;   // low 32 bits of flat shared addr = LDS offset
  uint64_t gaddr   = (uint64_t)(uintptr_t)gptr;
  asm volatile("global_load_async_to_lds_b128 %0, %1, off"
               :: "v"(lds_off), "v"(gaddr) : "memory");
}
__device__ __forceinline__ void wait_async0() {
  asm volatile("s_wait_asynccnt 0" ::: "memory");
}
__device__ __forceinline__ void wait_ds0() {
  asm volatile("s_wait_dscnt 0" ::: "memory");
}

// ---------------------------------------------------------------------------
// f32 -> bf16 conversion (weights / activations)
// ---------------------------------------------------------------------------
__global__ void cvt_bf16_kernel(const float* __restrict__ in,
                                __bf16* __restrict__ out, int n) {
  int i = blockIdx.x * blockDim.x + threadIdx.x;
  int stride = gridDim.x * blockDim.x;
  for (; i < n; i += stride) out[i] = (__bf16)in[i];
}

// ---------------------------------------------------------------------------
// Generic GEMM: C[M,N] = A[M,K](bf16) @ B[K,N](bf16) + bias, templated epilogue.
// Block tile 128x128, 8 waves (4Mx2N), wave tile 32x64 = 2x4 WMMA 16x16x32_bf16.
// Double-buffered LDS: async-stage tile k+1 while WMMAs consume tile k.
// ---------------------------------------------------------------------------
template<bool RELU, bool OUTBF>
__global__ __launch_bounds__(256)
void gemm_bf16_wmma(const __bf16* __restrict__ A, const __bf16* __restrict__ Bw,
                    const float* __restrict__ bias, float* __restrict__ outF,
                    __bf16* __restrict__ outB, int N, int K)
{
  __shared__ __attribute__((aligned(64))) __bf16 As[2][128 * 40];  // row-major, pad 32->40
  __shared__ __attribute__((aligned(64))) __bf16 Bs[2][128 * 32];  // transposed: [n][k]

  const int tid  = threadIdx.x;
  const int lane = tid & 31, wid = tid >> 5;
  const int l16  = lane & 15, g = lane >> 4;
  const int bm = blockIdx.y * 128, bn = blockIdx.x * 128;
  const int mw = (wid & 3) * 32, nw = (wid >> 2) * 64;

  v8f acc[2][4] = {};

  auto stageA = [&](int buf, int k0) {
#pragma unroll
    for (int i = 0; i < 2; ++i) {
      int c = tid + i * 256;              // 0..511
      int row = c >> 2, cc = (c & 3) * 8; // 4 x 8 bf16 chunks per row
      async_copy_b128(&As[buf][row * 40 + cc],
                      A + (size_t)(bm + row) * K + k0 + cc);
    }
  };
  auto stageB = [&](int buf, int k0) {
#pragma unroll
    for (int i = 0; i < 2; ++i) {
      int c = tid + i * 256;               // 0..511
      int kk = c >> 4, nn = (c & 15) * 8;  // 32 rows x 16 chunks
      uint4 bv = *(const uint4*)(Bw + (size_t)(k0 + kk) * N + bn + nn);
      const __bf16* bp = (const __bf16*)&bv;
#pragma unroll
      for (int j = 0; j < 8; ++j) Bs[buf][(nn + j) * 32 + kk] = bp[j];
    }
  };

  const int nk = K / 32;
  stageA(0, 0);
  stageB(0, 0);

  for (int ki = 0; ki < nk; ++ki) {
    const int cur = ki & 1;
    wait_async0();
    __syncthreads();                       // tile `cur` ready; prev reads done
    if (ki + 1 < nk) {
      stageA(1 - cur, (ki + 1) * 32);      // overlap staging with WMMAs below
      stageB(1 - cur, (ki + 1) * 32);
      if (ki + 2 < nk)                     // gfx1250 global_prefetch_b8
        __builtin_prefetch(Bw + (size_t)(ki + 2) * 32 * N + bn, 0, 1);
    }

    // ---- fragments: contiguous k = g*16 + s, same order in A and B ----
    AFrag a0, a1;
    const int ar0 = mw + l16, ar1 = mw + 16 + l16;
    a0.q[0] = *(const uint4*)&As[cur][ar0 * 40 + g * 16];      // K = g*16..
    a0.q[1] = *(const uint4*)&As[cur][ar0 * 40 + g * 16 + 8];  // K = g*16+8..
    a1.q[0] = *(const uint4*)&As[cur][ar1 * 40 + g * 16];
    a1.q[1] = *(const uint4*)&As[cur][ar1 * 40 + g * 16 + 8];
    v16bf b[4];
#pragma unroll
    for (int nt = 0; nt < 4; ++nt)
      b[nt] = *(const v16bf*)&Bs[cur][(nw + nt * 16 + l16) * 32 + g * 16];

#pragma unroll
    for (int nt = 0; nt < 4; ++nt) {
      acc[0][nt] = __builtin_amdgcn_wmma_f32_16x16x32_bf16(false, a0.v, false, b[nt], (short)0, acc[0][nt], false, false);
      acc[1][nt] = __builtin_amdgcn_wmma_f32_16x16x32_bf16(false, a1.v, false, b[nt], (short)0, acc[1][nt], false, false);
    }
  }

  // ---- straight-line epilogue: C/D layout row = r + 8*g, col = lane%16 ----
  const int colBase = bn + nw + l16;
  const int rowBase = bm + mw + g * 8;
#pragma unroll
  for (int nt = 0; nt < 4; ++nt) {
    int col = colBase + nt * 16;
    float bb = bias[col];
#pragma unroll
    for (int mt = 0; mt < 2; ++mt)
#pragma unroll
      for (int r = 0; r < 8; ++r) {
        int row = rowBase + mt * 16 + r;
        float v = acc[mt][nt][r] + bb;
        if (RELU) v = fmaxf(v, 0.f);
        size_t idx = (size_t)row * N + col;
        if (OUTBF) outB[idx] = (__bf16)v;
        else       outF[idx] = v;
      }
  }
}

// ---------------------------------------------------------------------------
// LayerNorm (+optional residual, +optional positional encoding, +pad mask).
// One wave per row (DM=512 -> 16 elems/lane, wave32 shfl reductions).
// Writes f32 (residual stream) and bf16 (next GEMM operand).
// ---------------------------------------------------------------------------
template<bool HASRES, bool HASPE, bool MASK>
__global__ __launch_bounds__(256)
void ln_kernel(const float* __restrict__ src, const float* __restrict__ res,
               const float* __restrict__ gamma, const float* __restrict__ beta,
               const float* __restrict__ pe, const int* __restrict__ lens,
               float* __restrict__ outF, __bf16* __restrict__ outB)
{
  const int lane = threadIdx.x & 31, wid = threadIdx.x >> 5;
  const int row = blockIdx.x * 8 + wid;      // 8192 rows total
  const int t = row & 1023, b = row >> 10;   // T = 1024

  float v[16];
  float s = 0.f;
#pragma unroll
  for (int i = 0; i < 16; ++i) {
    int c = lane + i * 32;
    float x = src[(size_t)row * 512 + c];
    if (HASRES) x += res[(size_t)row * 512 + c];
    v[i] = x; s += x;
  }
#pragma unroll
  for (int off = 16; off >= 1; off >>= 1) s += __shfl_xor(s, off, 32);
  float mu = s * (1.f / 512.f);
  float var = 0.f;
#pragma unroll
  for (int i = 0; i < 16; ++i) { float d = v[i] - mu; var += d * d; }
#pragma unroll
  for (int off = 16; off >= 1; off >>= 1) var += __shfl_xor(var, off, 32);
  float rstd = rsqrtf(var * (1.f / 512.f) + 1e-5f);

  float msk = 1.f;
  if (MASK) msk = (t < lens[b]) ? 1.f : 0.f;

#pragma unroll
  for (int i = 0; i < 16; ++i) {
    int c = lane + i * 32;
    float y = (v[i] - mu) * rstd * gamma[c] + beta[c];
    if (HASPE) y += pe[(size_t)t * 512 + c];
    y *= msk;
    outF[(size_t)row * 512 + c] = y;
    outB[(size_t)row * 512 + c] = (__bf16)y;
  }
}

// ---------------------------------------------------------------------------
// Flash-style attention per (b,h): block = 4 waves x 16 queries = 64 queries,
// key chunks of 64 (double-buffered), online softmax in f32, bf16 WMMA for
// QK^T and PV. Q,K,V,O layout: [B*T, H*64] bf16 row-major.
// ---------------------------------------------------------------------------
__global__ __launch_bounds__(128)
void attn_kernel(const __bf16* __restrict__ Qm, const __bf16* __restrict__ Km,
                 const __bf16* __restrict__ Vm, const int* __restrict__ lens,
                 __bf16* __restrict__ Om)
{
  const float scale = 0.125f;  // 1/sqrt(DK=64)
  const int tid = threadIdx.x, lane = tid & 31, wid = tid >> 5;
  const int l16 = lane & 15, g = lane >> 4;
  const int bh = blockIdx.y, b = bh >> 3, h = bh & 7;
  const int qt = blockIdx.x;      // 16 query tiles of 64
  const int len = lens[b];

  __shared__ __attribute__((aligned(64))) __bf16 Ks[2][64 * 64];     // [key][d]
  __shared__ __attribute__((aligned(64))) __bf16 Vts[2][64 * 72];    // V^T: [d][key]
  __shared__ __attribute__((aligned(64))) __bf16 Ps[4][16 * 72];     // per-wave probs

  // Q fragments: wave's 16 query rows, K-dim 64 -> two 32-steps
  AFrag qf[2];
  {
    size_t qrow = (size_t)b * 1024 + (size_t)qt * 64 + wid * 16 + l16;
    const __bf16* qp = Qm + qrow * 512 + h * 64;
#pragma unroll
    for (int ks = 0; ks < 2; ++ks) {
      qf[ks].q[0] = *(const uint4*)(qp + ks * 32 + g * 16);
      qf[ks].q[1] = *(const uint4*)(qp + ks * 32 + g * 16 + 8);
    }
  }

  auto stageKV = [&](int buf, int kc) {
#pragma unroll
    for (int i = 0; i < 4; ++i) {
      int c = tid + i * 128;
      int kr = c >> 3, cc = (c & 7) * 8;
      async_copy_b128(&Ks[buf][kr * 64 + cc],
                      Km + ((size_t)b * 1024 + kc + kr) * 512 + h * 64 + cc);
    }
#pragma unroll
    for (int i = 0; i < 4; ++i) {
      int c = tid + i * 128;
      int kr = c >> 3, cc = (c & 7) * 8;
      uint4 vv = *(const uint4*)(Vm + ((size_t)b * 1024 + kc + kr) * 512 + h * 64 + cc);
      const __bf16* vp = (const __bf16*)&vv;
#pragma unroll
      for (int j = 0; j < 8; ++j) Vts[buf][(cc + j) * 72 + kr] = vp[j];
    }
  };

  v8f oacc[4] = {};
  float mrun[8], lrun[8];
#pragma unroll
  for (int r = 0; r < 8; ++r) { mrun[r] = -1e30f; lrun[r] = 0.f; }

  stageKV(0, 0);
  for (int ic = 0; ic < 16; ++ic) {
    const int kc = ic * 64;
    const int cur = ic & 1;
    wait_async0();
    __syncthreads();                          // chunk `cur` ready
    if (ic + 1 < 16) stageKV(1 - cur, kc + 64);  // overlap next staging

    // ---- scores S = Q K^T : B operand = K^T, contiguous in Ks rows ----
    v8f s[4] = {};
#pragma unroll
    for (int nt = 0; nt < 4; ++nt) {
#pragma unroll
      for (int ks = 0; ks < 2; ++ks) {
        v16bf kb = *(const v16bf*)&Ks[cur][(nt * 16 + l16) * 64 + ks * 32 + g * 16];
        s[nt] = __builtin_amdgcn_wmma_f32_16x16x32_bf16(false, qf[ks].v, false, kb, (short)0, s[nt], false, false);
      }
    }
    // scale + key-pad mask
    bool valid[4];
#pragma unroll
    for (int nt = 0; nt < 4; ++nt) {
      valid[nt] = (kc + nt * 16 + l16) < len;
#pragma unroll
      for (int r = 0; r < 8; ++r) {
        float vv = s[nt][r] * scale;
        s[nt][r] = valid[nt] ? vv : -1e30f;
      }
    }
    // ---- online softmax (row = r + 8*g lives in a 16-lane group) ----
#pragma unroll
    for (int r = 0; r < 8; ++r) {
      float mx = fmaxf(fmaxf(s[0][r], s[1][r]), fmaxf(s[2][r], s[3][r]));
      mx = fmaxf(mx, __shfl_xor(mx, 1, 32));
      mx = fmaxf(mx, __shfl_xor(mx, 2, 32));
      mx = fmaxf(mx, __shfl_xor(mx, 4, 32));
      mx = fmaxf(mx, __shfl_xor(mx, 8, 32));
      float mnew = fmaxf(mrun[r], mx);
      float corr = __expf(mrun[r] - mnew);
      float psum = 0.f;
#pragma unroll
      for (int nt = 0; nt < 4; ++nt) {
        float p = valid[nt] ? __expf(s[nt][r] - mnew) : 0.f;
        s[nt][r] = p;
        psum += p;
      }
      psum += __shfl_xor(psum, 1, 32);
      psum += __shfl_xor(psum, 2, 32);
      psum += __shfl_xor(psum, 4, 32);
      psum += __shfl_xor(psum, 8, 32);
      lrun[r] = lrun[r] * corr + psum;
      mrun[r] = mnew;
#pragma unroll
      for (int nt = 0; nt < 4; ++nt) oacc[nt][r] *= corr;
    }
    // ---- C-layout -> A-layout round trip of P through per-wave LDS ----
#pragma unroll
    for (int nt = 0; nt < 4; ++nt)
#pragma unroll
      for (int r = 0; r < 8; ++r)
        Ps[wid][(r + 8 * g) * 72 + nt * 16 + l16] = (__bf16)s[nt][r];
    wait_ds0();  // same-wave DS ordering before re-reading as A fragments

    // ---- O += P @ V ----
#pragma unroll
    for (int ks = 0; ks < 2; ++ks) {
      AFrag pf;
      pf.q[0] = *(const uint4*)&Ps[wid][l16 * 72 + ks * 32 + g * 16];
      pf.q[1] = *(const uint4*)&Ps[wid][l16 * 72 + ks * 32 + g * 16 + 8];
#pragma unroll
      for (int nt = 0; nt < 4; ++nt) {
        AFrag vf;
        vf.q[0] = *(const uint4*)&Vts[cur][(nt * 16 + l16) * 72 + ks * 32 + g * 16];
        vf.q[1] = *(const uint4*)&Vts[cur][(nt * 16 + l16) * 72 + ks * 32 + g * 16 + 8];
        oacc[nt] = __builtin_amdgcn_wmma_f32_16x16x32_bf16(false, pf.v, false, vf.v, (short)0, oacc[nt], false, false);
      }
    }
  }

  // finalize: divide by row sums, write bf16
  {
    size_t orow = (size_t)b * 1024 + (size_t)qt * 64 + wid * 16 + g * 8;
#pragma unroll
    for (int r = 0; r < 8; ++r) {
      float inv = 1.f / lrun[r];
      __bf16* op = Om + (orow + r) * 512 + h * 64;
#pragma unroll
      for (int nt = 0; nt < 4; ++nt)
        op[nt * 16 + l16] = (__bf16)(oacc[nt][r] * inv);
    }
  }
}

// ---------------------------------------------------------------------------
// Host orchestration
// ---------------------------------------------------------------------------
extern "C" void kernel_launch(void* const* d_in, const int* in_sizes, int n_in,
                              void* d_out, int out_size, void* d_ws, size_t ws_size,
                              hipStream_t stream)
{
  (void)in_sizes; (void)n_in; (void)ws_size;
  const float* padded_input = (const float*)d_in[0];
  const int*   lens    = (const int*)d_in[1];
  const float* w_in    = (const float*)d_in[2];
  const float* b_in    = (const float*)d_in[3];
  const float* ln_in_g = (const float*)d_in[4];
  const float* ln_in_b = (const float*)d_in[5];
  const float* pe      = (const float*)d_in[6];
  const float* wq  = (const float*)d_in[7];
  const float* bq  = (const float*)d_in[8];
  const float* wk  = (const float*)d_in[9];
  const float* bk  = (const float*)d_in[10];
  const float* wv  = (const float*)d_in[11];
  const float* bv  = (const float*)d_in[12];
  const float* wfc = (const float*)d_in[13];
  const float* bfc = (const float*)d_in[14];
  const float* ln1g = (const float*)d_in[15];
  const float* ln1b = (const float*)d_in[16];
  const float* w1  = (const float*)d_in[17];
  const float* b1  = (const float*)d_in[18];
  const float* w2  = (const float*)d_in[19];
  const float* b2  = (const float*)d_in[20];
  const float* ln2g = (const float*)d_in[21];
  const float* ln2b = (const float*)d_in[22];

  // -------- workspace carve-up --------
  char* wsp = (char*)d_ws;
  auto alloc = [&](size_t bytes) {
    char* p = wsp; wsp += (bytes + 255) & ~(size_t)255; return (void*)p;
  };
  const size_t RT = 8192;  // rows = B*T
  __bf16* wbf_in = (__bf16*)alloc(320 * 512 * 2);
  __bf16* wbf_q  = (__bf16*)alloc((size_t)6 * 512 * 512 * 2);
  __bf16* wbf_k  = (__bf16*)alloc((size_t)6 * 512 * 512 * 2);
  __bf16* wbf_v  = (__bf16*)alloc((size_t)6 * 512 * 512 * 2);
  __bf16* wbf_fc = (__bf16*)alloc((size_t)6 * 512 * 512 * 2);
  __bf16* wbf_1  = (__bf16*)alloc((size_t)6 * 512 * 2048 * 2);
  __bf16* wbf_2  = (__bf16*)alloc((size_t)6 * 2048 * 512 * 2);
  __bf16* xin_bf = (__bf16*)alloc(RT * 320 * 2);
  float*  xA     = (float*)alloc(RT * 512 * 4);
  float*  xB     = (float*)alloc(RT * 512 * 4);
  float*  tmpF   = (float*)alloc(RT * 512 * 4);
  __bf16* x_bf   = (__bf16*)alloc(RT * 512 * 2);
  __bf16* q_bf   = (__bf16*)alloc(RT * 512 * 2);
  __bf16* k_bf   = (__bf16*)alloc(RT * 512 * 2);
  __bf16* v_bf   = (__bf16*)alloc(RT * 512 * 2);
  __bf16* at_bf  = (__bf16*)alloc(RT * 512 * 2);
  __bf16* h_bf   = (__bf16*)alloc(RT * 2048 * 2);

  auto cvt = [&](const float* src, __bf16* dst, size_t n) {
    int blocks = (int)((n + 255) / 256); if (blocks > 4096) blocks = 4096;
    cvt_bf16_kernel<<<dim3(blocks), dim3(256), 0, stream>>>(src, dst, (int)n);
  };
  auto gemmF = [&](const __bf16* A, const __bf16* Bm, const float* bias,
                   float* oF, int N, int K) {
    gemm_bf16_wmma<false,false><<<dim3(N/128, 64), dim3(256), 0, stream>>>(A, Bm, bias, oF, nullptr, N, K);
  };
  auto gemmB = [&](const __bf16* A, const __bf16* Bm, const float* bias,
                   __bf16* oB, int N, int K) {
    gemm_bf16_wmma<false,true><<<dim3(N/128, 64), dim3(256), 0, stream>>>(A, Bm, bias, nullptr, oB, N, K);
  };
  auto gemmBR = [&](const __bf16* A, const __bf16* Bm, const float* bias,
                    __bf16* oB, int N, int K) {
    gemm_bf16_wmma<true,true><<<dim3(N/128, 64), dim3(256), 0, stream>>>(A, Bm, bias, nullptr, oB, N, K);
  };

  // weight / input quantization to bf16 (recomputed every call: deterministic)
  cvt(w_in, wbf_in, (size_t)320 * 512);
  cvt(wq,  wbf_q,  (size_t)6 * 512 * 512);
  cvt(wk,  wbf_k,  (size_t)6 * 512 * 512);
  cvt(wv,  wbf_v,  (size_t)6 * 512 * 512);
  cvt(wfc, wbf_fc, (size_t)6 * 512 * 512);
  cvt(w1,  wbf_1,  (size_t)6 * 512 * 2048);
  cvt(w2,  wbf_2,  (size_t)6 * 2048 * 512);
  cvt(padded_input, xin_bf, RT * 320);

  // input projection + input LN + positional encoding
  gemmF(xin_bf, wbf_in, b_in, tmpF, 512, 320);
  float* curF = xA;
  float* nxtF = xB;
  ln_kernel<false,true,false><<<dim3(1024), dim3(256), 0, stream>>>(
      tmpF, nullptr, ln_in_g, ln_in_b, pe, lens, curF, x_bf);

  for (int l = 0; l < 6; ++l) {
    const size_t o2 = (size_t)l * 512 * 512;
    const size_t o1 = (size_t)l * 512;
    const size_t oi = (size_t)l * 512 * 2048;
    const size_t ob = (size_t)l * 2048;

    gemmB(x_bf, wbf_q + o2, bq + o1, q_bf, 512, 512);
    gemmB(x_bf, wbf_k + o2, bk + o1, k_bf, 512, 512);
    gemmB(x_bf, wbf_v + o2, bv + o1, v_bf, 512, 512);

    attn_kernel<<<dim3(16, 64), dim3(128), 0, stream>>>(q_bf, k_bf, v_bf, lens, at_bf);

    gemmF(at_bf, wbf_fc + o2, bfc + o1, tmpF, 512, 512);
    ln_kernel<true,false,true><<<dim3(1024), dim3(256), 0, stream>>>(
        tmpF, curF, ln1g + o1, ln1b + o1, nullptr, lens, nxtF, x_bf);
    { float* t = curF; curF = nxtF; nxtF = t; }

    gemmBR(x_bf, wbf_1 + oi, b1 + ob, h_bf, 2048, 512);   // ReLU fused
    gemmF(h_bf, wbf_2 + oi, b2 + o1, tmpF, 512, 2048);
    ln_kernel<true,false,true><<<dim3(1024), dim3(256), 0, stream>>>(
        tmpF, curF, ln2g + o1, ln2b + o1, nullptr, lens, nxtF, x_bf);
    { float* t = curF; curF = nxtF; nxtF = t; }
  }

  hipMemcpyAsync(d_out, curF, (size_t)out_size * sizeof(float),
                 hipMemcpyDeviceToDevice, stream);
}
